// SparseDynamicVoxelAttention_80547816669629
// MI455X (gfx1250) — matrix operand order repaired and checked by reading, not compiled, not run on hardware
//
#include <hip/hip_runtime.h>
#include <hip/hip_bf16.h>

// ---------------------------------------------------------------------------
// SparseDynamicVoxelAttention for MI455X (gfx1250, wave32, WMMA)
// B=4, V=4096, D=256, H=4, Dh=64, KNN=10, TK=4, HID=64
// ---------------------------------------------------------------------------

#define BB   4
#define VV   4096
#define DD   256
#define HH   4
#define DHH  64
#define KNNK 10
#define TKK_SEL 4
#define HIDN 64

typedef unsigned int u32;
typedef unsigned long long u64;
typedef __attribute__((ext_vector_type(4)))  u32    v4u;
typedef __attribute__((ext_vector_type(4)))  float  v4f;
typedef __attribute__((ext_vector_type(8)))  float  v8f;
typedef __attribute__((ext_vector_type(16))) __bf16 v16bf;

union Frag16  { v4u q[2]; v16bf v;     };   // 32 B = 8 VGPRs / lane
union Pack16  { v4u q[2]; __bf16 e[16]; };  // 16 converted elements

__device__ __forceinline__ u64 shflxor_u64(u64 x, int m) {
  u32 lo = (u32)x, hi = (u32)(x >> 32);
  lo = __shfl_xor(lo, m, 32);
  hi = __shfl_xor(hi, m, 32);
  return ((u64)hi << 32) | lo;
}

// ---------------------------------------------------------------------------
// Kernel 1: brute-force KNN (top-11 smallest d2, tie-break by index), one
// wave32 per query voxel. Keys = (bits(d2)<<32)|idx -> unsigned compare gives
// exactly jax.lax.top_k ordering (d2>=0 so float bits are monotone).
// ---------------------------------------------------------------------------
#define KNNP1 11
__global__ __launch_bounds__(256)
void knn_kernel(const float* __restrict__ coords, int* __restrict__ knnIdx)
{
  const int lane = threadIdx.x & 31;
  const int wv   = blockIdx.x * 8 + (threadIdx.x >> 5);   // 0..B*V-1
  const int b    = wv >> 12;
  const int v    = wv & (VV - 1);
  const float* cb = coords + (size_t)b * VV * 3;

  const float cvx = cb[v * 3 + 0], cvy = cb[v * 3 + 1], cvz = cb[v * 3 + 2];
  const float x2v = cvx * cvx + cvy * cvy + cvz * cvz;

  u64 keys[KNNP1];
#pragma unroll
  for (int i = 0; i < KNNP1; ++i) keys[i] = ~0ull;

  for (int u = lane; u < VV; u += 32) {
    float ux = cb[u * 3 + 0], uy = cb[u * 3 + 1], uz = cb[u * 3 + 2];
    float x2u = ux * ux + uy * uy + uz * uz;
    float dot = cvx * ux + cvy * uy + cvz * uz;
    float d2  = fmaxf(x2v + x2u - 2.0f * dot, 0.0f);   // reference formula
    u64 key = ((u64)__float_as_uint(d2) << 32) | (u32)u;
    if (key < keys[KNNP1 - 1]) {
      // branchless sorted insert (bubble the max out) - static indexing only
#pragma unroll
      for (int j = 0; j < KNNP1; ++j) {
        u64 kj = keys[j];
        bool lt = key < kj;
        keys[j] = lt ? key : kj;
        key     = lt ? kj  : key;
      }
    }
  }

  // Cross-lane merge: pop the global min 11 times (keys are unique).
#pragma unroll 1
  for (int r = 0; r < KNNP1; ++r) {
    u64 mine = keys[0];
    u64 g = mine;
#pragma unroll
    for (int m = 16; m >= 1; m >>= 1) {
      u64 o = shflxor_u64(g, m);
      g = (o < g) ? o : g;
    }
    if (mine == g) {               // winner pops its front element
#pragma unroll
      for (int j = 0; j < KNNP1 - 1; ++j) keys[j] = keys[j + 1];
      keys[KNNP1 - 1] = ~0ull;
    }
    if (r > 0 && lane == 0)        // rank 0 == self, dropped (knn_all[:,:,1:])
      knnIdx[(size_t)wv * KNNK + (r - 1)] = (int)(g & 0xFFFFFFFFu);
  }
}

// ---------------------------------------------------------------------------
// Kernel 2: generic bf16 WMMA GEMM  C[M,N] = X[M,K] * W[K,N] (+bias)(*mask)
// Block tile 128x64, 8 waves (4M x 2N), wave tile 32x32:
//   2 A-frags x 2 B-frags -> 4 v_wmma_f32_16x16x32_bf16 per K-step,
//   each fragment reused twice.  K is a template constant so the K-loop
//   fully unrolls (K=256 -> 8 steps -> 32 static WMMAs).
// Fragment layouts per CDNA5 ISA 7.12.2 (wave32).
// Requires: M%128==0, N%64==0 (true for all call sites here).
// ---------------------------------------------------------------------------
#define GT_M 128
#define GT_N 64
#define GT_K 32

template <int K>
__global__ __launch_bounds__(256)
void gemm_bf16_wmma(const float* __restrict__ X, int ldx,
                    const float* __restrict__ W, int ldw,
                    const float* __restrict__ bias,
                    const unsigned char* __restrict__ rowMask,
                    float* __restrict__ C, int ldc)
{
  __shared__ alignas(16) __bf16 ldsA[GT_M][40];   // [m][k], 80B row stride
  __shared__ alignas(16) __bf16 ldsBT[GT_N][40];  // [n][k] (B transposed)

  const int t    = threadIdx.x;
  const int lane = t & 31;
  const int wave = t >> 5;
  const int wm   = wave >> 1;         // 0..3  (M sub-tile, 32 rows each)
  const int wn   = wave & 1;          // 0..1  (N sub-tile, 32 cols each)
  const int rowBase = blockIdx.y * GT_M;
  const int colBase = blockIdx.x * GT_N;

  v8f acc00 = {}, acc01 = {}, acc10 = {}, acc11 = {};

  const int arow = t >> 1, acol = (t & 1) * 16;   // A: 128r x 32k, 16/thr
  const int bkr  = t >> 3, bc0  = (t & 7) * 8;    // B: 32k  x 64n,  8/thr

#pragma unroll
  for (int k0 = 0; k0 < K; k0 += GT_K) {
    __syncthreads();
    { // stage A tile: 16 fp32 -> bf16 (native cvt), 2x ds_store_b128
      const float* src = X + (size_t)(rowBase + arow) * ldx + k0 + acol;
      v4f f0 = *reinterpret_cast<const v4f*>(src);
      v4f f1 = *reinterpret_cast<const v4f*>(src + 4);
      v4f f2 = *reinterpret_cast<const v4f*>(src + 8);
      v4f f3 = *reinterpret_cast<const v4f*>(src + 12);
      Pack16 p;
#pragma unroll
      for (int j = 0; j < 4; ++j) {
        p.e[j]      = (__bf16)f0[j];
        p.e[4 + j]  = (__bf16)f1[j];
        p.e[8 + j]  = (__bf16)f2[j];
        p.e[12 + j] = (__bf16)f3[j];
      }
      v4u* d = reinterpret_cast<v4u*>(&ldsA[arow][acol]);
      d[0] = p.q[0];
      d[1] = p.q[1];
      if (k0 + GT_K < K) __builtin_prefetch(src + GT_K, 0, 0); // global_prefetch_b8
    }
    { // stage B tile transposed: ldsBT[n][k] = W[k0+k][colBase+n]
      const float* src = W + (size_t)(k0 + bkr) * ldw + colBase + bc0;
      v4f f0 = *reinterpret_cast<const v4f*>(src);
      v4f f1 = *reinterpret_cast<const v4f*>(src + 4);
#pragma unroll
      for (int j = 0; j < 4; ++j) ldsBT[bc0 + j][bkr]     = (__bf16)f0[j];
#pragma unroll
      for (int j = 0; j < 4; ++j) ldsBT[bc0 + 4 + j][bkr] = (__bf16)f1[j];
    }
    __syncthreads();

    // A fragments: lane<16 -> K 0-7 / 16-23 ; lane>=16 -> K 8-15 / 24-31
    Frag16 a0, a1, b0, b1;
    const int ar  = wm * 32 + (lane & 15);
    const int akb = (lane < 16) ? 0 : 8;
    a0.q[0] = *reinterpret_cast<const v4u*>(&ldsA[ar][akb]);
    a0.q[1] = *reinterpret_cast<const v4u*>(&ldsA[ar][akb + 16]);
    a1.q[0] = *reinterpret_cast<const v4u*>(&ldsA[ar + 16][akb]);
    a1.q[1] = *reinterpret_cast<const v4u*>(&ldsA[ar + 16][akb + 16]);
    // B fragments: lane<16 -> K 0-7 / 16-23 ; lane>=16 -> K 8-15 / 24-31
    const int bkb = (lane < 16) ? 0 : 8;
    const int c0  = wn * 32 + (lane & 15);
    b0.q[0] = *reinterpret_cast<const v4u*>(&ldsBT[c0][bkb]);
    b0.q[1] = *reinterpret_cast<const v4u*>(&ldsBT[c0][bkb + 16]);
    b1.q[0] = *reinterpret_cast<const v4u*>(&ldsBT[c0 + 16][bkb]);
    b1.q[1] = *reinterpret_cast<const v4u*>(&ldsBT[c0 + 16][bkb + 16]);

    acc00 = __builtin_amdgcn_wmma_f32_16x16x32_bf16(false, a0.v, false, b0.v,
                                                    (short)0, acc00, false, false);
    acc01 = __builtin_amdgcn_wmma_f32_16x16x32_bf16(false, a0.v, false, b1.v,
                                                    (short)0, acc01, false, false);
    acc10 = __builtin_amdgcn_wmma_f32_16x16x32_bf16(false, a1.v, false, b0.v,
                                                    (short)0, acc10, false, false);
    acc11 = __builtin_amdgcn_wmma_f32_16x16x32_bf16(false, a1.v, false, b1.v,
                                                    (short)0, acc11, false, false);
  }

  // Epilogue. C layout: VGPR r, lane l -> M = r + (l>=16 ? 8 : 0), N = l&15.
  const int mBase = rowBase + wm * 32 + ((lane >> 4) << 3);
  const int n0    = colBase + wn * 32 + (lane & 15);
  const float bia0 = bias ? bias[n0]      : 0.0f;
  const float bia1 = bias ? bias[n0 + 16] : 0.0f;
#pragma unroll
  for (int r = 0; r < 8; ++r) {
    const int m0 = mBase + r;        // A-frag 0 rows
    const int m1 = m0 + 16;          // A-frag 1 rows
    const float sc0 = rowMask ? (rowMask[m0] ? 1.0f : 0.0f) : 1.0f;
    const float sc1 = rowMask ? (rowMask[m1] ? 1.0f : 0.0f) : 1.0f;
    C[(size_t)m0 * ldc + n0]      = (acc00[r] + bia0) * sc0;
    C[(size_t)m0 * ldc + n0 + 16] = (acc01[r] + bia1) * sc0;
    C[(size_t)m1 * ldc + n0]      = (acc10[r] + bia0) * sc1;
    C[(size_t)m1 * ldc + n0 + 16] = (acc11[r] + bia1) * sc1;
  }
}

// ---------------------------------------------------------------------------
// Kernel 3: per-edge scoring + top-4 neighbor selection. One wave per voxel,
// 2 hidden dims per lane.  h = relu(Ea[v] + Eb[n] + rel_pos @ W1c); score=h.W2
// softmax is monotone -> top_k(softmax(s)) == top_k(s): skip softmax.
// ---------------------------------------------------------------------------
__global__ __launch_bounds__(256)
void edge_kernel(const float* __restrict__ coords,
                 const float* __restrict__ Ea,   // (B*V,64)  tokens@W1a + b1
                 const float* __restrict__ Eb,   // (B*V,64)  tokens@W1b
                 const float* __restrict__ W1,   // (515,64)
                 const float* __restrict__ W2,   // (64,1)
                 const float* __restrict__ b2,   // (1,)
                 const int*   __restrict__ knnIdx,
                 int*         __restrict__ absIdx)
{
  const int lane = threadIdx.x & 31;
  const int wv   = blockIdx.x * 8 + (threadIdx.x >> 5);
  const int b    = wv >> 12;
  const int v    = wv & (VV - 1);
  const float* cb = coords + (size_t)b * VV * 3;

  const float cvx = cb[v * 3 + 0], cvy = cb[v * 3 + 1], cvz = cb[v * 3 + 2];

  int knn[KNNK];
#pragma unroll
  for (int k = 0; k < KNNK; ++k) knn[k] = knnIdx[(size_t)wv * KNNK + k];

  const int j0 = lane * 2, j1 = lane * 2 + 1;
  const float ea0 = Ea[(size_t)wv * HIDN + j0];
  const float ea1 = Ea[(size_t)wv * HIDN + j1];
  const float w2a = W2[j0], w2b = W2[j1];
  const float wcx0 = W1[(2 * DD + 0) * HIDN + j0], wcx1 = W1[(2 * DD + 0) * HIDN + j1];
  const float wcy0 = W1[(2 * DD + 1) * HIDN + j0], wcy1 = W1[(2 * DD + 1) * HIDN + j1];
  const float wcz0 = W1[(2 * DD + 2) * HIDN + j0], wcz1 = W1[(2 * DD + 2) * HIDN + j1];
  const float b2v = b2[0];

  float sc[KNNK];
#pragma unroll 1
  for (int k = 0; k < KNNK; ++k) {
    const int n  = knn[k];
    const int ng = b * VV + n;
    const float rpx = cb[n * 3 + 0] - cvx;
    const float rpy = cb[n * 3 + 1] - cvy;
    const float rpz = cb[n * 3 + 2] - cvz;
    float h0 = ea0 + Eb[(size_t)ng * HIDN + j0] + rpx * wcx0 + rpy * wcy0 + rpz * wcz0;
    float h1 = ea1 + Eb[(size_t)ng * HIDN + j1] + rpx * wcx1 + rpy * wcy1 + rpz * wcz1;
    h0 = fmaxf(h0, 0.0f);
    h1 = fmaxf(h1, 0.0f);
    float p = h0 * w2a + h1 * w2b;
#pragma unroll
    for (int m = 16; m >= 1; m >>= 1) p += __shfl_xor(p, m, 32);
    sc[k] = p + b2v;
  }

  // top-4 by descending score, ties -> smaller local index (jax top_k order)
  u64 kk[KNNK];
#pragma unroll
  for (int k = 0; k < KNNK; ++k) {
    u32 bts = __float_as_uint(sc[k]);
    u32 asc = bts ^ ((bts >> 31) ? 0xFFFFFFFFu : 0x80000000u); // ascending map
    u32 dsc = ~asc;                                            // descending map
    kk[k] = ((u64)dsc << 32) | (u32)k;
  }
  int sel[TKK_SEL];
#pragma unroll
  for (int t4 = 0; t4 < TKK_SEL; ++t4) {
    u64 mn = ~0ull;
#pragma unroll
    for (int k = 0; k < KNNK; ++k) mn = (kk[k] < mn) ? kk[k] : mn;
    sel[t4] = (int)(mn & 0xFFu);
#pragma unroll
    for (int k = 0; k < KNNK; ++k) if (kk[k] == mn) kk[k] = ~0ull;
  }
  if (lane == 0) {
#pragma unroll
    for (int t4 = 0; t4 < TKK_SEL; ++t4)
      absIdx[(size_t)wv * TKK_SEL + t4] = knn[sel[t4]];
  }
}

// ---------------------------------------------------------------------------
// Kernel 4: per-voxel 4-key attention. One wave per voxel; lane l owns dims
// [8l, 8l+8) so head h = lanes [8h, 8h+8); intra-head reduce via shfl_xor.
// Writes attn_out in-place over the Q buffer (row-private).
// ---------------------------------------------------------------------------
__global__ __launch_bounds__(256)
void attn_kernel(float* __restrict__ Qbuf,
                 const float* __restrict__ Kbuf,
                 const float* __restrict__ Vbuf,
                 const int* __restrict__ absIdx)
{
  const int lane = threadIdx.x & 31;
  const int wv   = blockIdx.x * 8 + (threadIdx.x >> 5);
  const int b    = wv >> 12;

  const size_t qoff = (size_t)wv * DD + lane * 8;
  const v4f q0 = *reinterpret_cast<const v4f*>(Qbuf + qoff);
  const v4f q1 = *reinterpret_cast<const v4f*>(Qbuf + qoff + 4);

  int nrow[TKK_SEL];
#pragma unroll
  for (int k = 0; k < TKK_SEL; ++k)
    nrow[k] = b * VV + absIdx[(size_t)wv * TKK_SEL + k];

  float scr[TKK_SEL];
#pragma unroll
  for (int k = 0; k < TKK_SEL; ++k) {
    const float* kr = Kbuf + (size_t)nrow[k] * DD + lane * 8;
    const v4f k0 = *reinterpret_cast<const v4f*>(kr);
    const v4f k1 = *reinterpret_cast<const v4f*>(kr + 4);
    float p = q0[0] * k0[0] + q0[1] * k0[1] + q0[2] * k0[2] + q0[3] * k0[3]
            + q1[0] * k1[0] + q1[1] * k1[1] + q1[2] * k1[2] + q1[3] * k1[3];
    p += __shfl_xor(p, 1, 32);
    p += __shfl_xor(p, 2, 32);
    p += __shfl_xor(p, 4, 32);          // full dot over this head's 64 dims
    scr[k] = p * 0.125f;                // / sqrt(Dh), Dh = 64
  }

  float mx = scr[0];
#pragma unroll
  for (int k = 1; k < TKK_SEL; ++k) mx = fmaxf(mx, scr[k]);
  float e[TKK_SEL]; float s = 0.0f;
#pragma unroll
  for (int k = 0; k < TKK_SEL; ++k) { e[k] = expf(scr[k] - mx); s += e[k]; }
  const float inv = 1.0f / s;

  v4f o0 = {}; v4f o1 = {};
#pragma unroll
  for (int k = 0; k < TKK_SEL; ++k) {
    const float* vr = Vbuf + (size_t)nrow[k] * DD + lane * 8;
    const v4f v0 = *reinterpret_cast<const v4f*>(vr);
    const v4f v1 = *reinterpret_cast<const v4f*>(vr + 4);
    const float w = e[k] * inv;
    o0 += w * v0;
    o1 += w * v1;
  }
  *reinterpret_cast<v4f*>(Qbuf + qoff)     = o0;
  *reinterpret_cast<v4f*>(Qbuf + qoff + 4) = o1;
}

// ---------------------------------------------------------------------------
// Host launcher
// ---------------------------------------------------------------------------
extern "C" void kernel_launch(void* const* d_in, const int* in_sizes, int n_in,
                              void* d_out, int out_size, void* d_ws, size_t ws_size,
                              hipStream_t stream)
{
  const float*         tokens = (const float*)d_in[0];
  const float*         coords = (const float*)d_in[1];
  const unsigned char* mask   = (const unsigned char*)d_in[2];  // numpy bool
  const float* W1 = (const float*)d_in[3];
  const float* b1 = (const float*)d_in[4];
  const float* W2 = (const float*)d_in[5];
  const float* b2 = (const float*)d_in[6];
  const float* Wq = (const float*)d_in[7];
  const float* bq = (const float*)d_in[8];
  const float* Wk = (const float*)d_in[9];
  const float* bk = (const float*)d_in[10];
  const float* Wv = (const float*)d_in[11];
  const float* bv = (const float*)d_in[12];
  const float* Wo = (const float*)d_in[13];
  const float* bo = (const float*)d_in[14];
  float* out = (float*)d_out;

  const int ROWS = BB * VV;                       // 16384
  char* ws = (char*)d_ws;
  int*   knnIdx = (int*)(ws + 0);                 //   655,360 B
  int*   absIdx = (int*)(ws + 655360);            //   262,144 B
  float* Ea     = (float*)(ws + (1u  << 20));     // 4,194,304 B
  float* Eb     = (float*)(ws + 5242880u);        // 4,194,304 B
  float* Qb     = (float*)(ws + 9437184u);        // 16 MB (later: attn_out)
  float* Kb     = (float*)(ws + 26214400u);       // 16 MB
  float* Vb     = (float*)(ws + 42991616u);       // 16 MB  (end ~57 MB)

  const dim3 blk(256);
  const dim3 gVox(ROWS / 8);                      // wave-per-voxel kernels
  const dim3 gN64(1, ROWS / GT_M);                // N=64  GEMMs
  const dim3 gN256(DD / GT_N, ROWS / GT_M);       // N=256 GEMMs

  // 1) KNN (top-10 neighbors, self excluded)
  knn_kernel<<<gVox, blk, 0, stream>>>(coords, knnIdx);

  // 2) edge-MLP decomposition: Ea = X@W1[0:256] + b1 ; Eb = X@W1[256:512]
  gemm_bf16_wmma<DD><<<gN64, blk, 0, stream>>>(tokens, DD, W1,             HIDN, b1,      nullptr, Ea, HIDN);
  gemm_bf16_wmma<DD><<<gN64, blk, 0, stream>>>(tokens, DD, W1 + DD * HIDN, HIDN, nullptr, nullptr, Eb, HIDN);

  // 3) edge scores + top-4 neighbor selection
  edge_kernel<<<gVox, blk, 0, stream>>>(coords, Ea, Eb, W1, W2, b2, knnIdx, absIdx);

  // 4) Q/K/V projections (projection commutes with gather)
  gemm_bf16_wmma<DD><<<gN256, blk, 0, stream>>>(tokens, DD, Wq, DD, bq, nullptr, Qb, DD);
  gemm_bf16_wmma<DD><<<gN256, blk, 0, stream>>>(tokens, DD, Wk, DD, bk, nullptr, Kb, DD);
  gemm_bf16_wmma<DD><<<gN256, blk, 0, stream>>>(tokens, DD, Wv, DD, bv, nullptr, Vb, DD);

  // 5) 4-key multi-head attention (in-place Qb -> attn_out)
  attn_kernel<<<gVox, blk, 0, stream>>>(Qb, Kb, Vb, absIdx);

  // 6) output projection + bias + mask
  gemm_bf16_wmma<DD><<<gN256, blk, 0, stream>>>(Qb, DD, Wo, DD, bo, mask, out, DD);
}
